// TrajectoryPredictor_7052336300580
// MI455X (gfx1250) — matrix operand order repaired and checked by reading, not compiled, not run on hardware
//
#include <hip/hip_runtime.h>
#include <math.h>

#define B 4096
#define T 20
#define STATE_DIM 10
#define HIDDEN 256
#define LATENT 64
#define GRU_IN 320
#define DEG2RAD 0.01745329252f

typedef __attribute__((ext_vector_type(16))) _Float16 v16h;
typedef __attribute__((ext_vector_type(8)))  _Float16 v8h;
typedef __attribute__((ext_vector_type(8)))  float    v8f;

__device__ __forceinline__ float sigmoidf_(float x){ return 1.f/(1.f+__expf(-x)); }
// ELU: exp(x)-1 for x<0 lowers to a single v_exp_f32 (branch-free, unlike expm1f's
// divergent polynomial slow path).
__device__ __forceinline__ float eluf_(float x){ return x>0.f ? x : (__expf(x)-1.f); }

__device__ __forceinline__ v16h cat16(v8h lo, v8h hi){
    return __builtin_shufflevector(lo, hi, 0,1,2,3,4,5,6,7,8,9,10,11,12,13,14,15);
}

// ---------------------------------------------------------------------------
// WMMA GEMM:  C(MxN) = act( A(MxK) @ W(NxK)^T + bias )
// Block 128 threads = 4 waves; block tile 64x64, K step 32; each wave owns a
// 32x32 (2x2 WMMA 16x16x32 f16) micro-tile. Double-buffered LDS: per K-step,
// global loads for tile k+1 are issued first, WMMAs consume tile k, converted
// data is stored to the other buffer, then a single barrier.
// M%64==0, N%64==0, K%32==0 (all GEMMs in this model satisfy this).
// ---------------------------------------------------------------------------
#define BM 64
#define BN 64
#define BK 32
#define LSTR 40   // LDS row stride in halves (80B: keeps 16B alignment, staggers banks)

__device__ __forceinline__ void cvt_store16(const float4* f, _Float16* dst){
    v8h h0, h1;
    h0[0]=(_Float16)f[0].x; h0[1]=(_Float16)f[0].y; h0[2]=(_Float16)f[0].z; h0[3]=(_Float16)f[0].w;
    h0[4]=(_Float16)f[1].x; h0[5]=(_Float16)f[1].y; h0[6]=(_Float16)f[1].z; h0[7]=(_Float16)f[1].w;
    h1[0]=(_Float16)f[2].x; h1[1]=(_Float16)f[2].y; h1[2]=(_Float16)f[2].z; h1[3]=(_Float16)f[2].w;
    h1[4]=(_Float16)f[3].x; h1[5]=(_Float16)f[3].y; h1[6]=(_Float16)f[3].z; h1[7]=(_Float16)f[3].w;
    *(v8h*)dst = h0; *(v8h*)(dst+8) = h1;
}

__global__ void __launch_bounds__(128) gemm_bias_act(
    const float* __restrict__ A, int lda,
    const float* __restrict__ W, int ldw,
    const float* __restrict__ bias,
    float* __restrict__ C, int ldc,
    int K, int act)
{
    __shared__ __align__(16) _Float16 Ash[2][BM*LSTR];
    __shared__ __align__(16) _Float16 Bsh[2][BN*LSTR];

    const int tid  = threadIdx.x;
    const int lane = tid & 31;
    const int wave = tid >> 5;
    const size_t m0 = (size_t)blockIdx.y * BM;
    const size_t n0 = (size_t)blockIdx.x * BN;
    const int wm = (wave >> 1) * 32;
    const int wn = (wave & 1) * 32;
    const int lrow = tid >> 1;           // 0..63
    const int lcol = (tid & 1) * 16;     // 0 or 16

    v8f acc[2][2];
    #pragma unroll
    for (int i=0;i<2;i++)
        #pragma unroll
        for (int j=0;j<2;j++)
            #pragma unroll
            for (int r=0;r<8;r++) acc[i][j][r] = 0.f;

    const float* Aptr = A + (m0 + lrow)*(size_t)lda + lcol;
    const float* Wptr = W + (n0 + lrow)*(size_t)ldw + lcol;
    const int soff = lrow*LSTR + lcol;

    // preload + stage tile 0 into buffer 0
    float4 fa[4], fb[4];
    {
        const float4* sa = (const float4*)(Aptr);
        const float4* sb = (const float4*)(Wptr);
        #pragma unroll
        for (int q=0;q<4;q++){ fa[q]=sa[q]; fb[q]=sb[q]; }
        cvt_store16(fa, &Ash[0][soff]);
        cvt_store16(fb, &Bsh[0][soff]);
    }
    __syncthreads();

    const int nk = K / BK;
    const int lg = lane >> 4;   // 0/1
    const int lr = lane & 15;

    for (int kt = 0; kt < nk; kt++) {
        const int cur = kt & 1;
        // issue global loads for the next tile first (latency hidden by WMMAs)
        if (kt + 1 < nk) {
            const float4* sa = (const float4*)(Aptr + (kt+1)*BK);
            const float4* sb = (const float4*)(Wptr + (kt+1)*BK);
            #pragma unroll
            for (int q=0;q<4;q++){ fa[q]=sa[q]; fb[q]=sb[q]; }
        }
        // compute from current buffer
        v16h af[2], bf[2];
        #pragma unroll
        for (int i=0;i<2;i++){
            // A 16x32 f16 layout: lane-half lg selects the K interleave of 8
            const _Float16* p = &Ash[cur][(wm + i*16 + lr)*LSTR];
            v8h lo = *(const v8h*)(p + 8*lg);        // K = 8*lg .. +7
            v8h hi = *(const v8h*)(p + 16 + 8*lg);   // K = 16+8*lg .. +7
            af[i] = cat16(lo, hi);
        }
        #pragma unroll
        for (int j=0;j<2;j++){
            // B 32x16 f16 layout: lane-half lg selects the K interleave of 8
            const _Float16* p = &Bsh[cur][(wn + j*16 + lr)*LSTR + 8*lg];
            v8h lo = *(const v8h*)(p);
            v8h hi = *(const v8h*)(p + 16);
            bf[j] = cat16(lo, hi);
        }
        #pragma unroll
        for (int i=0;i<2;i++)
            #pragma unroll
            for (int j=0;j<2;j++)
                acc[i][j] = __builtin_amdgcn_wmma_f32_16x16x32_f16(
                    false, af[i], false, bf[j], (short)0, acc[i][j], false, false);
        // stage next tile into the other buffer; single barrier per iteration
        if (kt + 1 < nk) {
            cvt_store16(fa, &Ash[cur^1][soff]);
            cvt_store16(fb, &Bsh[cur^1][soff]);
            __syncthreads();
        }
    }

    #pragma unroll
    for (int i=0;i<2;i++){
        #pragma unroll
        for (int j=0;j<2;j++){
            const size_t n = n0 + wn + j*16 + lr;
            const float bv = bias[n];
            #pragma unroll
            for (int r=0;r<8;r++){
                size_t m = m0 + wm + i*16 + lg*8 + r;
                float v = acc[i][j][r] + bv;
                if (act) v = eluf_(v);
                C[m*(size_t)ldc + n] = v;
            }
        }
    }
}

// ---------------------------------------------------------------------------
// State encoder: deg_sc(10->12) -> Linear(12,128) ELU -> Linear(128,128) ELU
// One block (128 threads) per batch row; writes X cols [0,128).
// ---------------------------------------------------------------------------
__global__ void __launch_bounds__(128) state_encoder(
    const float* __restrict__ past, const float* __restrict__ future, int t,
    const float* __restrict__ w1, const float* __restrict__ b1,
    const float* __restrict__ w2, const float* __restrict__ b2,
    float* __restrict__ X)
{
    const int b = blockIdx.x;
    const int j = threadIdx.x;
    __shared__ float raw[10];
    __shared__ float in12[12];
    __shared__ float h[128];
    const float* s = (t < 8) ? (past   + (size_t)b*8*STATE_DIM  + t*STATE_DIM)
                             : (future + (size_t)b*12*STATE_DIM + (t-8)*STATE_DIM);
    if (j < 10) raw[j] = s[j];
    __syncthreads();
    if (j < 12) {
        float v;
        if      (j <  4) v = raw[j];
        else if (j == 4) v = __sinf(raw[4]*DEG2RAD);
        else if (j == 5) v = __cosf(raw[4]*DEG2RAD);
        else if (j <  9) v = raw[j-1];
        else if (j == 9) v = __sinf(raw[8]*DEG2RAD);
        else if (j ==10) v = __cosf(raw[8]*DEG2RAD);
        else             v = raw[9];
        in12[j] = v;
    }
    __syncthreads();
    float a = b1[j];
    #pragma unroll
    for (int k=0;k<12;k++) a += in12[k]*w1[j*12+k];
    h[j] = eluf_(a);
    __syncthreads();
    float o = b2[j];
    for (int k=0;k<128;k++) o += h[k]*w2[j*128+k];
    X[(size_t)b*GRU_IN + j] = eluf_(o);
}

// ---------------------------------------------------------------------------
// World encoder: Linear(16,64) ELU -> Linear(64,64) ELU ; writes X cols [256,320)
// ---------------------------------------------------------------------------
__global__ void __launch_bounds__(64) world_encoder(
    const float* __restrict__ world, int t,
    const float* __restrict__ w1, const float* __restrict__ b1,
    const float* __restrict__ w2, const float* __restrict__ b2,
    float* __restrict__ X)
{
    const int b = blockIdx.x;
    const int j = threadIdx.x;
    __shared__ float in16[16];
    __shared__ float h[64];
    const float* w = world + ((size_t)b*T + t)*16;
    if (j < 16) in16[j] = w[j];
    __syncthreads();
    float a = b1[j];
    #pragma unroll
    for (int k=0;k<16;k++) a += in16[k]*w1[j*16+k];
    h[j] = eluf_(a);
    __syncthreads();
    float o = b2[j];
    for (int k=0;k<64;k++) o += h[k]*w2[j*64+k];
    X[(size_t)b*GRU_IN + 256 + j] = eluf_(o);
}

// ---------------------------------------------------------------------------
// Neighbor attention: 2 heads x 32 dims over 8 neighbors + out-proj + LayerNorm.
// Block = 64 threads = 2 waves; wave == head (score reduce via shfl_xor).
// Reads se from X cols [0,128); writes X cols [128,192).
// ---------------------------------------------------------------------------
__global__ void __launch_bounds__(64) neighbor_attn(
    const float* __restrict__ ctxs, int t,
    const float* __restrict__ qw, const float* __restrict__ qb,
    const float* __restrict__ kw, const float* __restrict__ kb,
    const float* __restrict__ vw, const float* __restrict__ vb,
    const float* __restrict__ ow, const float* __restrict__ ob,
    const float* __restrict__ lng, const float* __restrict__ lnb,
    float* __restrict__ X)
{
    const int b = blockIdx.x;
    const int j = threadIdx.x;  // 0..63 output/feature dim
    __shared__ float se[128];
    __shared__ float ctx[8][8];
    __shared__ float nrm2[8];
    __shared__ float osh[64];
    __shared__ float psh[64];

    se[j]      = X[(size_t)b*GRU_IN + j];
    se[j + 64] = X[(size_t)b*GRU_IN + 64 + j];
    const float* cb = ctxs + (((size_t)b*T + t)*8)*8;
    ctx[j>>3][j&7] = cb[j];
    __syncthreads();
    if (j < 8) {
        float ns = 0.f;
        #pragma unroll
        for (int c=0;c<8;c++) ns += ctx[j][c]*ctx[j][c];
        nrm2[j] = ns;
    }
    float q = qb[j];
    for (int k=0;k<128;k++) q += se[k]*qw[j*128+k];
    float Kv[8], Vv[8];
    #pragma unroll
    for (int i=0;i<8;i++){
        float kk = kb[j], vv = vb[j];
        #pragma unroll
        for (int c=0;c<8;c++){ float cc = ctx[i][c]; kk += cc*kw[j*8+c]; vv += cc*vw[j*8+c]; }
        Kv[i]=kk; Vv[i]=vv;
    }
    __syncthreads();
    float sc[8];
    #pragma unroll
    for (int i=0;i<8;i++){
        float p = q*Kv[i];
        #pragma unroll
        for (int off=16; off>0; off>>=1) p += __shfl_xor(p, off, 32);
        p *= 0.17677669529f;  // 1/sqrt(32)
        sc[i] = (nrm2[i] < 1e-12f) ? -1e9f : p;
    }
    float mx = sc[0];
    #pragma unroll
    for (int i=1;i<8;i++) mx = fmaxf(mx, sc[i]);
    float den = 0.f;
    #pragma unroll
    for (int i=0;i<8;i++){ sc[i] = __expf(sc[i]-mx); den += sc[i]; }
    float o = 0.f;
    #pragma unroll
    for (int i=0;i<8;i++) o += (sc[i]/den) * Vv[i];
    osh[j] = o;
    __syncthreads();
    float op = ob[j];
    for (int k=0;k<64;k++) op += osh[k]*ow[j*64+k];
    psh[j] = op;
    __syncthreads();
    float mu = 0.f;
    for (int k=0;k<64;k++) mu += psh[k];
    mu *= (1.f/64.f);
    float var = 0.f;
    for (int k=0;k<64;k++){ float d = psh[k]-mu; var += d*d; }
    var *= (1.f/64.f);
    X[(size_t)b*GRU_IN + 128 + j] = (op - mu) * rsqrtf(var + 1e-5f) * lng[j] + lnb[j];
}

// ---------------------------------------------------------------------------
__global__ void gru_combine(const float* __restrict__ gi, const float* __restrict__ gh,
                            float* __restrict__ H)
{
    int idx = blockIdx.x*blockDim.x + threadIdx.x;
    if (idx >= B*HIDDEN) return;
    int b = idx / HIDDEN, u = idx % HIDDEN;
    const float* gir = gi + (size_t)b*3*HIDDEN;
    const float* ghr = gh + (size_t)b*3*HIDDEN;
    float r  = sigmoidf_(gir[u] + ghr[u]);
    float zg = sigmoidf_(gir[HIDDEN+u] + ghr[HIDDEN+u]);
    float n  = tanhf(gir[2*HIDDEN+u] + r*ghr[2*HIDDEN+u]);
    H[idx] = (1.f - zg)*n + zg*H[idx];
}

__global__ void pack_post_dec(const float* __restrict__ H1, const float* __restrict__ X,
                              float* __restrict__ PostIn, float* __restrict__ DecIn)
{
    int idx = blockIdx.x*blockDim.x + threadIdx.x;
    if (idx >= B*384) return;
    int b = idx / 384, u = idx % 384;
    float v;
    if (u < 256) { v = H1[(size_t)b*HIDDEN + u]; DecIn[(size_t)b*GRU_IN + u] = v; }
    else         { v = X[(size_t)b*GRU_IN + (u-256)]; }
    PostIn[(size_t)b*384 + u] = v;
}

__global__ void klz_kernel(const float* __restrict__ post, const float* __restrict__ prior,
                           const float* __restrict__ noise, int t,
                           float* __restrict__ DecIn, float* __restrict__ X,
                           float* __restrict__ out_kl)
{
    int b = blockIdx.x*blockDim.x + threadIdx.x;
    if (b >= B) return;
    const float* po  = post  + (size_t)b*128;   // pm | pls
    const float* pr  = prior + (size_t)b*128;   // qm | qls
    const float* eps = noise + ((size_t)t*B + b)*LATENT;
    float kl = 0.f;
    for (int l=0;l<LATENT;l++){
        float pm  = po[l];
        float pls = fminf(fmaxf(po[64+l], -5.f), 2.f);
        float qm  = pr[l];
        float qls = fminf(fmaxf(pr[64+l], -5.f), 2.f);
        float var1 = __expf(2.f*pls), var2 = __expf(2.f*qls);
        float d = qm - pm;
        kl += 0.5f*(var1/var2 + d*d/var2 - 1.f + 2.f*(qls - pls));
        float z = pm + __expf(pls)*eps[l];
        DecIn[(size_t)b*GRU_IN + 256 + l] = z;
        X[(size_t)b*GRU_IN + 192 + l] = z;     // z for next step's GRU input
    }
    out_kl[(size_t)b*T + t] = kl;
}

__constant__ float cCLAMP[10] = {0.05f,0.05f,3333.f,200.f,180.f,10000.f,200.f,0.3f,360.f,60.f};

__global__ void dec_final(const float* __restrict__ H,
                          const float* __restrict__ w3, const float* __restrict__ b3,
                          int t, float* __restrict__ out_pred)
{
    int idx = blockIdx.x*blockDim.x + threadIdx.x;
    if (idx >= B*STATE_DIM) return;
    int b = idx / STATE_DIM, s = idx % STATE_DIM;
    const float* hr = H + (size_t)b*128;
    float a = b3[s];
    for (int k=0;k<128;k++) a += hr[k]*w3[s*128+k];
    float c = cCLAMP[s];
    a = fminf(fmaxf(a, -c), c);
    out_pred[((size_t)b*(T-1) + t)*STATE_DIM + s] = a;
}

__global__ void target_kernel(const float* __restrict__ past, const float* __restrict__ future,
                              float* __restrict__ out_tgt)
{
    int idx = blockIdx.x*blockDim.x + threadIdx.x;
    if (idx >= B*(T-1)*STATE_DIM) return;
    int s  = idx % STATE_DIM;
    int bt = idx / STATE_DIM;
    int t  = bt % (T-1);
    int b  = bt / (T-1);
    float s0 = (t   < 8) ? past[(size_t)b*80 + t*10 + s]     : future[(size_t)b*120 + (t-8)*10 + s];
    float s1 = (t+1 < 8) ? past[(size_t)b*80 + (t+1)*10 + s] : future[(size_t)b*120 + (t-7)*10 + s];
    out_tgt[idx] = s1 - s0;
}

__global__ void init_kernel(float* __restrict__ H0, float* __restrict__ H1, float* __restrict__ X)
{
    int idx = blockIdx.x*blockDim.x + threadIdx.x;
    if (idx < B*HIDDEN) { H0[idx] = 0.f; H1[idx] = 0.f; }
    if (idx < B*LATENT) X[(size_t)(idx/LATENT)*GRU_IN + 192 + (idx%LATENT)] = 0.f;
}

// ---------------------------------------------------------------------------
extern "C" void kernel_launch(void* const* d_in, const int* in_sizes, int n_in,
                              void* d_out, int out_size, void* d_ws, size_t ws_size,
                              hipStream_t stream)
{
    (void)in_sizes; (void)n_in; (void)out_size; (void)ws_size;
    const float* past   = (const float*)d_in[0];
    const float* future = (const float*)d_in[1];
    const float* ctxs   = (const float*)d_in[2];
    const float* world  = (const float*)d_in[3];
    const float* noise  = (const float*)d_in[4];
    const float* const* P = (const float* const*)(d_in + 5);
    const float *se_w1=P[0], *se_b1=P[1], *se_w2=P[2], *se_b2=P[3];
    const float *q_w=P[4], *q_b=P[5], *k_w=P[6], *k_b=P[7], *v_w=P[8], *v_b=P[9];
    const float *o_w=P[10], *o_b=P[11], *ln_g=P[12], *ln_b=P[13];
    const float *we_w1=P[14], *we_b1=P[15], *we_w2=P[16], *we_b2=P[17];
    const float *wih0=P[18], *whh0=P[19], *bih0=P[20], *bhh0=P[21];
    const float *wih1=P[22], *whh1=P[23], *bih1=P[24], *bhh1=P[25];
    const float *post_w1=P[26], *post_b1=P[27], *post_w2=P[28], *post_b2=P[29];
    const float *prior_w1=P[30], *prior_b1=P[31], *prior_w2=P[32], *prior_b2=P[33];
    const float *dec_w1=P[34], *dec_b1=P[35], *dec_w2=P[36], *dec_b2=P[37];
    const float *dec_w3=P[38], *dec_b3=P[39];

    char* ws = (char*)d_ws;
    size_t off = 0;
    auto alloc = [&](size_t bytes)->float* {
        float* p = (float*)(ws + off);
        off += (bytes + 255) & ~(size_t)255;
        return p;
    };
    float* X        = alloc((size_t)B*GRU_IN*4);    // [se|inter|z|we]
    float* H0       = alloc((size_t)B*HIDDEN*4);
    float* H1       = alloc((size_t)B*HIDDEN*4);
    float* Gi       = alloc((size_t)B*3*HIDDEN*4);
    float* Gh       = alloc((size_t)B*3*HIDDEN*4);
    float* PostIn   = alloc((size_t)B*384*4);
    float* HidA     = alloc((size_t)B*256*4);
    float* HidB     = alloc((size_t)B*256*4);
    float* PostOut  = alloc((size_t)B*128*4);
    float* PriorOut = alloc((size_t)B*128*4);
    float* DecIn    = alloc((size_t)B*GRU_IN*4);    // [h1|z]
    float* DecHid   = alloc((size_t)B*256*4);
    float* DecHid2  = alloc((size_t)B*128*4);

    float* out_pred = (float*)d_out;                              // (B, T-1, 10)
    float* out_tgt  = out_pred + (size_t)B*(T-1)*STATE_DIM;       // (B, T-1, 10)
    float* out_kl   = out_tgt  + (size_t)B*(T-1)*STATE_DIM;       // (B, T)

    dim3 b64(64), b128(128), b256(256);

    init_kernel<<<dim3((B*HIDDEN+255)/256), b256, 0, stream>>>(H0, H1, X);
    target_kernel<<<dim3((B*(T-1)*STATE_DIM+255)/256), b256, 0, stream>>>(past, future, out_tgt);

    for (int t = 0; t < T; ++t) {
        state_encoder<<<dim3(B), b128, 0, stream>>>(past, future, t, se_w1, se_b1, se_w2, se_b2, X);
        neighbor_attn<<<dim3(B), b64, 0, stream>>>(ctxs, t, q_w,q_b,k_w,k_b,v_w,v_b,o_w,o_b,ln_g,ln_b, X);
        world_encoder<<<dim3(B), b64, 0, stream>>>(world, t, we_w1,we_b1,we_w2,we_b2, X);

        // GRU layer 0
        gemm_bias_act<<<dim3(768/64, B/64), b128, 0, stream>>>(X,  GRU_IN, wih0, GRU_IN, bih0, Gi, 3*HIDDEN, GRU_IN, 0);
        gemm_bias_act<<<dim3(768/64, B/64), b128, 0, stream>>>(H0, HIDDEN, whh0, HIDDEN, bhh0, Gh, 3*HIDDEN, HIDDEN, 0);
        gru_combine<<<dim3((B*HIDDEN+255)/256), b256, 0, stream>>>(Gi, Gh, H0);
        // GRU layer 1
        gemm_bias_act<<<dim3(768/64, B/64), b128, 0, stream>>>(H0, HIDDEN, wih1, HIDDEN, bih1, Gi, 3*HIDDEN, HIDDEN, 0);
        gemm_bias_act<<<dim3(768/64, B/64), b128, 0, stream>>>(H1, HIDDEN, whh1, HIDDEN, bhh1, Gh, 3*HIDDEN, HIDDEN, 0);
        gru_combine<<<dim3((B*HIDDEN+255)/256), b256, 0, stream>>>(Gi, Gh, H1);

        pack_post_dec<<<dim3((B*384+255)/256), b256, 0, stream>>>(H1, X, PostIn, DecIn);
        // posterior
        gemm_bias_act<<<dim3(256/64, B/64), b128, 0, stream>>>(PostIn, 384, post_w1, 384, post_b1, HidA, 256, 384, 1);
        gemm_bias_act<<<dim3(128/64, B/64), b128, 0, stream>>>(HidA, 256, post_w2, 256, post_b2, PostOut, 128, 256, 0);
        // prior
        gemm_bias_act<<<dim3(256/64, B/64), b128, 0, stream>>>(H1, HIDDEN, prior_w1, 256, prior_b1, HidB, 256, 256, 1);
        gemm_bias_act<<<dim3(128/64, B/64), b128, 0, stream>>>(HidB, 256, prior_w2, 256, prior_b2, PriorOut, 128, 256, 0);

        klz_kernel<<<dim3((B+127)/128), b128, 0, stream>>>(PostOut, PriorOut, noise, t, DecIn, X, out_kl);

        if (t < T-1) {  // pred_deltas[:, :-1] drops the last step's delta
            gemm_bias_act<<<dim3(256/64, B/64), b128, 0, stream>>>(DecIn, GRU_IN, dec_w1, GRU_IN, dec_b1, DecHid, 256, GRU_IN, 1);
            gemm_bias_act<<<dim3(128/64, B/64), b128, 0, stream>>>(DecHid, 256, dec_w2, 256, dec_b2, DecHid2, 128, 256, 1);
            dec_final<<<dim3((B*STATE_DIM+255)/256), b256, 0, stream>>>(DecHid2, dec_w3, dec_b3, t, out_pred);
        }
    }
}
